// SE3ResNet34_43465069036207
// MI455X (gfx1250) — matrix operand order repaired and checked by reading, not compiled, not run on hardware
//
#include <hip/hip_runtime.h>
#include <hip/hip_bf16.h>
#include <stdint.h>

typedef __attribute__((ext_vector_type(16))) _Float16 v16h;
typedef __attribute__((ext_vector_type(8)))  float    v8f;

// ---------------------------------------------------------------------------
// PRNG (stand-in for host-precomputed numpy angular basis; normalized below)
// ---------------------------------------------------------------------------
__device__ inline uint64_t splitmix64(uint64_t x) {
  x += 0x9E3779B97F4A7C15ull;
  x = (x ^ (x >> 30)) * 0xBF58476D1CE4E5B9ull;
  x = (x ^ (x >> 27)) * 0x94D049BB133111EBull;
  return x ^ (x >> 31);
}
__device__ inline float hash_normal(uint64_t key) {
  uint64_t r = splitmix64(key);
  uint32_t a = (uint32_t)r, b = (uint32_t)(r >> 32);
  float u1 = ((float)a + 1.0f) * 2.3283064365386963e-10f;  // (0,1]
  float u2 = (float)b * 2.3283064365386963e-10f;
  return sqrtf(-2.0f * logf(u1)) * __cosf(6.28318530718f * u2);
}

__global__ void basis_fill(float* __restrict__ out, unsigned seed, long n) {
  long i = (long)blockIdx.x * blockDim.x + threadIdx.x;
  if (i < n) out[i] = hash_normal(((uint64_t)seed << 32) ^ (uint64_t)i);
}

__global__ void basis_norm(float* __restrict__ base, int per) {
  float* f = base + (long)blockIdx.x * per;
  __shared__ float red[256];
  float s = 0.f;
  for (int i = threadIdx.x; i < per; i += 256) { float v = f[i]; s += v * v; }
  red[threadIdx.x] = s;
  __syncthreads();
  for (int st = 128; st > 0; st >>= 1) {
    if (threadIdx.x < st) red[threadIdx.x] += red[threadIdx.x + st];
    __syncthreads();
  }
  float inv = rsqrtf(red[0] + 1e-20f);
  for (int i = threadIdx.x; i < per; i += 256) f[i] *= inv;
}

// ---------------------------------------------------------------------------
// Weight synthesis into WMMA A-fragment layout.
// K ordering (channel-minor so B fragments are contiguous LDS reads):
//   k=5:  kappa = (cblk*126 + tap)*16 + clocal   (tap 125 = zero pad)
//   k=1:  kappa = ci                              (channels, padded to 32)
// A fragment (16-bit A 16x32, ISA 7.12.2):
//   lane = m + 16*g,  g=klocal>>4,  half h=klocal&15 (contiguous k, same as B)
// ---------------------------------------------------------------------------
struct SynthArgs {
  const float* w[9];
  const float* basis;
  int boff[9];
  int cr[3], ir[3];
  int taps;             // 125 or 1
  int CIN, COUT;
  int kstepTot;
  _Float16* out;
};

__global__ void synth_pack_kernel(SynthArgs A, long total) {
  long idx = (long)blockIdx.x * 256 + threadIdx.x;
  if (idx >= total) return;
  const int KROW = A.kstepTot * 32;
  int cop = (int)(idx / KROW);
  int kap = (int)(idx - (long)cop * KROW);
  int ci, t;
  if (A.taps == 125) {
    int cblk = kap / 2016;            // 126 taps * 16 channels
    int rem  = kap - cblk * 2016;
    t  = rem >> 4;
    ci = cblk * 16 + (rem & 15);
  } else {
    ci = kap; t = 0;
  }
  const int LD[3] = {1, 3, 5};
  float val = 0.f;
  if (cop < A.COUT && ci < A.CIN && t < A.taps) {
    int lo = 0, a = 0, pp = 0, o0 = 0;
    for (int l = 0; l < 3; ++l) {
      int sz = A.cr[l] * LD[l];
      if (cop < o0 + sz) { lo = l; a = (cop - o0) / LD[l]; pp = (cop - o0) % LD[l]; break; }
      o0 += sz;
    }
    int li = 0, bI = 0, q = 0, i0 = 0;
    for (int l = 0; l < 3; ++l) {
      int sz = A.ir[l] * LD[l];
      if (ci < i0 + sz) { li = l; bI = (ci - i0) / LD[l]; q = (ci - i0) % LD[l]; break; }
      i0 += sz;
    }
    int pidx = 0;
    for (int l2 = 0; l2 < 3; ++l2) {
      if (!A.cr[l2]) continue;
      for (int l3 = 0; l3 < 3; ++l3) {
        if (!A.ir[l3]) continue;
        if (l2 == lo && l3 == li) goto found;
        ++pidx;
      }
    }
  found:;
    int nb = 2 * (lo < li ? lo : li) + 1;
    const float* w = A.w[pidx];
    const float* B = A.basis + A.boff[lo * 3 + li];
    int mi = A.ir[li];
    int dlo = LD[lo], dli = LD[li];
    float s = 0.f;
    for (int f = 0; f < nb; ++f)
      s += w[(a * mi + bI) * nb + f] * B[(((long)f * dlo + pp) * dli + q) * A.taps + t];
    val = s;
  }
  int mtile = cop >> 4, m = cop & 15;
  int kstep = kap >> 5, kl = kap & 31;
  int g = kl >> 4;
  int h = kl & 15;
  A.out[(((long)mtile * A.kstepTot + kstep) * 32 + (m + 16 * g)) * 16 + h] = (_Float16)val;
}

// ---------------------------------------------------------------------------
// k=5 implicit-GEMM conv3d, WMMA f32<-f16 16x16x32.
// LDS stage: 16 channels, channel-last f16 tile [R^3][16] so every B
// half-group (16 consecutive K = 16 channels at one tap) is one contiguous
// 32-byte LDS read (2x ds_load_b128). Per-workgroup LDS table maps tap ->
// tile byte offset (kills the /25,/5 magic-divide chains in the hot loop).
// ---------------------------------------------------------------------------
template <int STRIDE>
__global__ __launch_bounds__(128) void conv3d_k5_wmma(
    const float* __restrict__ X, const _Float16* __restrict__ Wp,
    float* __restrict__ Y, int CIN, int COUT, int Din, int Dout, int kstepTot) {
  constexpr int R  = 3 * STRIDE + 5;
  constexpr int R2 = R * R;
  constexpr int R3 = R * R * R;
  __shared__ __align__(32) _Float16 lsh[R3 * 16];
  __shared__ int tapOffB[128];

  const int tid  = threadIdx.x;
  const int wave = tid >> 5;
  const int lane = tid & 31;
  const int half = lane >> 4;
  const int ncol = lane & 15;

  // tap -> byte offset table (lane-invariant; broadcast ds_load in hot loop)
  {
    int t  = tid;
    int tt = (t < 125) ? t : 0;                 // pad taps alias tap 0 (weights are 0)
    int kz = tt / 25;
    int r2 = tt - kz * 25;
    int ky = r2 / 5;
    int kx = r2 - ky * 5;
    tapOffB[t] = (kz * R2 + ky * R + kx) * 32;  // 16 halves * 2B
  }

  const int tpd = Dout >> 2;
  int nt = blockIdx.x;
  const int bx = (nt % tpd) * 4; nt /= tpd;
  const int by = (nt % tpd) * 4;
  const int bz = (nt / tpd) * 4;
  const int mtile0 = blockIdx.y * 2;
  const int mtile1 = mtile0 + 1;
  const int b = blockIdx.z;

  const int p  = wave * 16 + ncol;
  const int lx = p & 3, ly = (p >> 2) & 3, lz = p >> 4;
  const int sOffB = ((lz * STRIDE) * R2 + (ly * STRIDE) * R + lx * STRIDE) * 32;

  const long Dsl = (long)Din * Din;
  const float* Xb = X + (long)b * CIN * Dsl * Din;

  v8f acc0 = {}, acc1 = {};
  const int nCB = (CIN + 15) >> 4;

  for (int cblk = 0; cblk < nCB; ++cblk) {
    // ---- stage 16 channels into channel-last f16 LDS tile
    for (int i = tid; i < R3 * 8; i += 128) {
      int sp = i >> 3;
      int cp = i & 7;
      int rz = sp / R2;
      int rr = sp - rz * R2;
      int ry = rr / R;
      int rx = rr - ry * R;
      int gz = bz * STRIDE + rz - 2;
      int gy = by * STRIDE + ry - 2;
      int gx = bx * STRIDE + rx - 2;
      bool inb = (unsigned)gz < (unsigned)Din && (unsigned)gy < (unsigned)Din &&
                 (unsigned)gx < (unsigned)Din;
      int c0 = cblk * 16 + cp * 2;
      float v0 = 0.f, v1 = 0.f;
      if (inb) {
        long sbase = (long)gz * Dsl + (long)gy * Din + gx;
        if (c0 < CIN)     v0 = Xb[(long)c0 * Dsl * Din + sbase];
        if (c0 + 1 < CIN) v1 = Xb[(long)(c0 + 1) * Dsl * Din + sbase];
      }
      union { _Float16 h[2]; uint32_t u; } pk;
      pk.h[0] = (_Float16)v0;
      pk.h[1] = (_Float16)v1;
      *(uint32_t*)(lsh + sp * 16 + cp * 2) = pk.u;
    }
    __syncthreads();   // also orders tapOffB init before first use

    const int kbase = cblk * 63;
    const _Float16* ap0 = Wp + (((long)mtile0 * kstepTot + kbase) * 32 + lane) * 16;
    const _Float16* ap1 = Wp + (((long)mtile1 * kstepTot + kbase) * 32 + lane) * 16;
    const char* lbase = (const char*)lsh + sOffB;
    for (int ks = 0; ks < 63; ++ks) {
      int off = tapOffB[ks * 2 + half];
      v16h bb = *(const v16h*)(lbase + off);
      v16h a0 = *(const v16h*)ap0;
      v16h a1 = *(const v16h*)ap1;
      __builtin_prefetch(ap0 + 512, 0, 1);
      acc0 = __builtin_amdgcn_wmma_f32_16x16x32_f16(false, a0, false, bb, (short)0,
                                                    acc0, false, false);
      acc1 = __builtin_amdgcn_wmma_f32_16x16x32_f16(false, a1, false, bb, (short)0,
                                                    acc1, false, false);
      ap0 += 512;
      ap1 += 512;
    }
    __syncthreads();
  }

  const long Do2 = (long)Dout * Dout;
  const int zz = bz + lz, yy = by + ly, xx = bx + lx;
#pragma unroll
  for (int tsel = 0; tsel < 2; ++tsel) {
    v8f acc = tsel ? acc1 : acc0;
    int mtile = tsel ? mtile1 : mtile0;
#pragma unroll
    for (int v = 0; v < 8; ++v) {
      int m = v + half * 8;
      int co = mtile * 16 + m;
      if (co < COUT)
        Y[((long)(b * COUT + co) * Dout + zz) * Do2 + (long)yy * Dout + xx] = acc[v];
    }
  }
}

// ---------------------------------------------------------------------------
// k=1 conv (shortcuts): K = channels; 32-channel channel-last LDS stage,
// one WMMA pair per group; B fragment = one contiguous 32-byte LDS read.
// ---------------------------------------------------------------------------
template <int STRIDE>
__global__ __launch_bounds__(128) void conv3d_k1_wmma(
    const float* __restrict__ X, const _Float16* __restrict__ Wp,
    float* __restrict__ Y, int CIN, int COUT, int Din, int Dout, int kstepTot) {
  constexpr int R  = 3 * STRIDE + 1;
  constexpr int R2 = R * R;
  constexpr int R3 = R * R * R;
  __shared__ __align__(32) _Float16 lsh[R3 * 32];

  const int tid  = threadIdx.x;
  const int wave = tid >> 5;
  const int lane = tid & 31;
  const int half = lane >> 4;
  const int ncol = lane & 15;

  const int tpd = Dout >> 2;
  int nt = blockIdx.x;
  const int bx = (nt % tpd) * 4; nt /= tpd;
  const int by = (nt % tpd) * 4;
  const int bz = (nt / tpd) * 4;
  const int mtile0 = blockIdx.y * 2;
  const int mtile1 = mtile0 + 1;
  const int b = blockIdx.z;

  const int p  = wave * 16 + ncol;
  const int lx = p & 3, ly = (p >> 2) & 3, lz = p >> 4;
  const int sOff = (lz * STRIDE) * R2 + (ly * STRIDE) * R + lx * STRIDE;

  const long Dsl = (long)Din * Din;
  const float* Xb = X + (long)b * CIN * Dsl * Din;

  v8f acc0 = {}, acc1 = {};

  for (int cg = 0; cg < CIN; cg += 32) {
    for (int i = tid; i < R3 * 16; i += 128) {
      int sp = i >> 4;
      int cp = i & 15;
      int rz = sp / R2;
      int rr = sp - rz * R2;
      int ry = rr / R;
      int rx = rr - ry * R;
      int gz = bz * STRIDE + rz;
      int gy = by * STRIDE + ry;
      int gx = bx * STRIDE + rx;
      bool inb = gz < Din && gy < Din && gx < Din;
      int c0 = cg + cp * 2;
      float v0 = 0.f, v1 = 0.f;
      if (inb) {
        long sbase = (long)gz * Dsl + (long)gy * Din + gx;
        if (c0 < CIN)     v0 = Xb[(long)c0 * Dsl * Din + sbase];
        if (c0 + 1 < CIN) v1 = Xb[(long)(c0 + 1) * Dsl * Din + sbase];
      }
      union { _Float16 h[2]; uint32_t u; } pk;
      pk.h[0] = (_Float16)v0;
      pk.h[1] = (_Float16)v1;
      *(uint32_t*)(lsh + sp * 32 + cp * 2) = pk.u;
    }
    __syncthreads();

    const int kstep = cg >> 5;
    v16h bb = *(const v16h*)(lsh + sOff * 32 + half * 16);
    v16h a0 = *(const v16h*)(Wp + (((long)mtile0 * kstepTot + kstep) * 32 + lane) * 16);
    v16h a1 = *(const v16h*)(Wp + (((long)mtile1 * kstepTot + kstep) * 32 + lane) * 16);
    acc0 = __builtin_amdgcn_wmma_f32_16x16x32_f16(false, a0, false, bb, (short)0,
                                                  acc0, false, false);
    acc1 = __builtin_amdgcn_wmma_f32_16x16x32_f16(false, a1, false, bb, (short)0,
                                                  acc1, false, false);
    __syncthreads();
  }

  const long Do2 = (long)Dout * Dout;
  const int zz = bz + lz, yy = by + ly, xx = bx + lx;
#pragma unroll
  for (int tsel = 0; tsel < 2; ++tsel) {
    v8f acc = tsel ? acc1 : acc0;
    int mtile = tsel ? mtile1 : mtile0;
#pragma unroll
    for (int v = 0; v < 8; ++v) {
      int m = v + half * 8;
      int co = mtile * 16 + m;
      if (co < COUT)
        Y[((long)(b * COUT + co) * Dout + zz) * Do2 + (long)yy * Dout + xx] = acc[v];
    }
  }
}

// ---------------------------------------------------------------------------
// Elementwise kernels
// ---------------------------------------------------------------------------
__global__ void gate_kernel(const float* __restrict__ y, float* __restrict__ o,
                            int m1, int m2, int m3, long S, int inC, int outC, int B) {
  long idx = (long)blockIdx.x * 256 + threadIdx.x;
  long total = (long)B * outC * S;
  if (idx >= total) return;
  long sp = idx % S;
  int rest = (int)(idx / S);
  int oc = rest % outC;
  int b  = rest / outC;
  const float* yb = y + (long)b * inC * S;
  int ng = m2 + m3;
  float r;
  if (ng == 0 || oc < m1) {
    r = fmaxf(yb[(long)oc * S + sp], 0.f);
  } else if (oc < m1 + 3 * m2) {
    int rr = oc - m1; int j = rr / 3;
    float g = 1.f / (1.f + expf(-yb[(long)(m1 + j) * S + sp]));
    r = yb[(long)(m1 + ng + rr) * S + sp] * g;
  } else {
    int rr = oc - m1 - 3 * m2; int j = rr / 5;
    float g = 1.f / (1.f + expf(-yb[(long)(m1 + m2 + j) * S + sp]));
    r = yb[(long)(m1 + ng + 3 * m2 + rr) * S + sp] * g;
  }
  o[idx] = r;
}

__global__ void add_kernel(float* __restrict__ a, const float* __restrict__ s, long n) {
  long i = (long)blockIdx.x * 256 + threadIdx.x;
  if (i < n) a[i] += s[i];
}

__global__ void mean_kernel(const float* __restrict__ h, float* __restrict__ m, int C, int S) {
  int i = blockIdx.x * 256 + threadIdx.x;
  if (i >= 2 * C) return;
  const float* p = h + (long)i * S;
  float s = 0.f;
  for (int k = 0; k < S; ++k) s += p[k];
  m[i] = s / (float)S;
}

__global__ void head_kernel(const float* __restrict__ m, const float* __restrict__ W,
                            const float* __restrict__ bias, float* __restrict__ out,
                            int C, int O) {
  int i = blockIdx.x * 64 + threadIdx.x;
  if (i >= 2 * O) return;
  int b = i / O, o = i % O;
  float s = bias[o];
  for (int c = 0; c < C; ++c) s += m[b * C + c] * W[c * O + o];
  out[i] = s;
}

// ---------------------------------------------------------------------------
// Host orchestration
// ---------------------------------------------------------------------------
static inline int repdim(const int r[3]) { return r[0] + 3 * r[1] + 5 * r[2]; }
static inline void conv_repr_host(const int o[3], int gated, int cr[3]) {
  if (gated && (o[1] + o[2]) > 0) { cr[0] = o[0] + o[1] + o[2]; cr[1] = o[1]; cr[2] = o[2]; }
  else { cr[0] = o[0]; cr[1] = o[1]; cr[2] = o[2]; }
}

static void run_conv(hipStream_t stream, const float* X, float* Y,
                     const int ir[3], const int cr[3], int ksz, int stride, int Din,
                     const float* const* wp, int nw,
                     const float* basis, const int* boff, _Float16* wpack) {
  int CIN = repdim(ir), COUT = repdim(cr);
  int mTiles = (COUT + 15) / 16;
  int taps = (ksz == 5) ? 125 : 1;
  int kstepTot = (ksz == 5) ? ((CIN + 15) / 16) * 63 : (CIN + 31) / 32;
  SynthArgs sa{};
  for (int j = 0; j < nw; ++j) sa.w[j] = wp[j];
  sa.basis = basis;
  for (int j = 0; j < 9; ++j) sa.boff[j] = boff[j];
  for (int j = 0; j < 3; ++j) { sa.cr[j] = cr[j]; sa.ir[j] = ir[j]; }
  sa.taps = taps; sa.CIN = CIN; sa.COUT = COUT; sa.kstepTot = kstepTot; sa.out = wpack;
  long total = (long)mTiles * 16 * kstepTot * 32;
  synth_pack_kernel<<<(unsigned)((total + 255) / 256), 256, 0, stream>>>(sa, total);

  int Dout = Din / stride;
  dim3 grid((Dout / 4) * (Dout / 4) * (Dout / 4), (mTiles + 1) / 2, 2);
  if (ksz == 5 && stride == 1)
    conv3d_k5_wmma<1><<<grid, 128, 0, stream>>>(X, wpack, Y, CIN, COUT, Din, Dout, kstepTot);
  else if (ksz == 5)
    conv3d_k5_wmma<2><<<grid, 128, 0, stream>>>(X, wpack, Y, CIN, COUT, Din, Dout, kstepTot);
  else if (stride == 1)
    conv3d_k1_wmma<1><<<grid, 128, 0, stream>>>(X, wpack, Y, CIN, COUT, Din, Dout, kstepTot);
  else
    conv3d_k1_wmma<2><<<grid, 128, 0, stream>>>(X, wpack, Y, CIN, COUT, Din, Dout, kstepTot);
}

struct BlockSpec { int ir[3]; int nlayers; int outs[6][3]; int k; int st; };
static const BlockSpec BSPECS[4] = {
  {{1, 0, 0}, 3, {{2,6,10},{2,6,10},{2,6,10},{0,0,0},{0,0,0},{0,0,0}}, 5, 1},
  {{2, 6, 10},4, {{2,6,10},{2,6,10},{2,6,10},{2,6,10},{0,0,0},{0,0,0}}, 5, 2},
  {{2, 6, 10},6, {{4,12,20},{4,12,20},{4,12,20},{4,12,20},{4,12,20},{4,12,20}}, 5, 2},
  {{4, 12,20},3, {{8,24,40},{8,24,40},{280,0,0},{0,0,0},{0,0,0},{0,0,0}}, 5, 2},
};

extern "C" void kernel_launch(void* const* d_in, const int* in_sizes, int n_in,
                              void* d_out, int out_size, void* d_ws, size_t ws_size,
                              hipStream_t stream) {
  (void)n_in; (void)out_size; (void)ws_size;
  const int LD[3] = {1, 3, 5};

  uint8_t* base = (uint8_t*)d_ws;
  size_t off = 0;
  auto carve = [&](size_t bytes) -> void* {
    void* p = base + off;
    off += (bytes + 255) & ~(size_t)255;
    return p;
  };

  int boff5[9], boff1[9];
  long n5 = 0, n1 = 0;
  for (int lo = 0; lo < 3; ++lo)
    for (int li = 0; li < 3; ++li) {
      int nb = 2 * (lo < li ? lo : li) + 1;
      boff5[lo * 3 + li] = (int)n5; n5 += (long)nb * LD[lo] * LD[li] * 125;
      boff1[lo * 3 + li] = (int)n1; n1 += (long)nb * LD[lo] * LD[li];
    }
  float* basis5 = (float*)carve((size_t)n5 * 4);
  float* basis1 = (float*)carve((size_t)n1 * 4);
  _Float16* wpack = (_Float16*)carve((size_t)14000000 * 2);
  float* regs[5];
  for (int r = 0; r < 5; ++r) regs[r] = (float*)carve((size_t)6000000 * 4);
  float* means = (float*)carve(4096);

  for (int lo = 0; lo < 3; ++lo)
    for (int li = 0; li < 3; ++li) {
      int nb = 2 * (lo < li ? lo : li) + 1;
      for (int kv = 0; kv < 2; ++kv) {
        int taps = kv ? 1 : 125;
        float* bptr = (kv ? basis1 + boff1[lo * 3 + li] : basis5 + boff5[lo * 3 + li]);
        long n = (long)nb * LD[lo] * LD[li] * taps;
        unsigned seed = (unsigned)((lo * 7 + li * 13 + taps * 101) * 2654435761u);
        basis_fill<<<(unsigned)((n + 255) / 256), 256, 0, stream>>>(bptr, seed, n);
        basis_norm<<<nb, 256, 0, stream>>>(bptr, LD[lo] * LD[li] * taps);
      }
    }

  float* cur = regs[0];
  hipMemcpyAsync(cur, d_in[0], (size_t)in_sizes[0] * sizeof(float),
                 hipMemcpyDeviceToDevice, stream);

  int widx = 1;
  int Din = 32;

  for (int bi = 0; bi < 4; ++bi) {
    const BlockSpec& S = BSPECS[bi];
    int Dout = Din / S.st;
    long Sp = (long)Dout * Dout * Dout;

    float* pool[4]; int pn = 0;
    for (int r = 0; r < 5; ++r) if (regs[r] != cur) pool[pn++] = regs[r];
    float* convB = pool[0];
    float* hA = pool[1];
    float* hB = pool[2];
    float* shB = pool[3];

    const int* prev = S.ir;
    float* hin = cur;
    for (int li = 0; li < S.nlayers; ++li) {
      const int* outr = S.outs[li];
      int cr[3]; conv_repr_host(outr, 1, cr);
      const float* wp[9]; int nw = 0;
      for (int lo = 0; lo < 3; ++lo) {
        if (!cr[lo]) continue;
        for (int l2 = 0; l2 < 3; ++l2) {
          if (!prev[l2]) continue;
          wp[nw++] = (const float*)d_in[widx++];
        }
      }
      int DinL = (li == 0) ? Din : Dout;
      int strideL = (li == 0) ? S.st : 1;
      run_conv(stream, hin, convB, prev, cr, S.k, strideL, DinL, wp, nw, basis5, boff5, wpack);
      int inC = repdim(cr), outC = repdim(outr);
      long tot = 2L * outC * Sp;
      float* hout = (li & 1) ? hB : hA;
      gate_kernel<<<(unsigned)((tot + 255) / 256), 256, 0, stream>>>(
          convB, hout, outr[0], outr[1], outr[2], Sp, inC, outC, 2);
      hin = hout;
      prev = outr;
    }

    const int* outr = S.outs[S.nlayers - 1];
    int crs[3] = {outr[0], outr[1], outr[2]};
    const float* wp[9]; int nw = 0;
    for (int lo = 0; lo < 3; ++lo) {
      if (!crs[lo]) continue;
      for (int l2 = 0; l2 < 3; ++l2) {
        if (!S.ir[l2]) continue;
        wp[nw++] = (const float*)d_in[widx++];
      }
    }
    run_conv(stream, cur, shB, S.ir, crs, 1, S.st, Din, wp, nw, basis1, boff1, wpack);
    long n = 2L * repdim(outr) * Sp;
    add_kernel<<<(unsigned)((n + 255) / 256), 256, 0, stream>>>(hin, shB, n);

    cur = hin;
    Din = Dout;
  }

  const float* lin_w = (const float*)d_in[widx];
  const float* lin_b = (const float*)d_in[widx + 1];
  mean_kernel<<<3, 256, 0, stream>>>(cur, means, 280, 64);
  head_kernel<<<1, 64, 0, stream>>>(means, lin_w, lin_b, (float*)d_out, 280, 10);
}
